// DiscreteEncoder_75634374082603
// MI455X (gfx1250) — matrix operand order repaired and checked by reading, not compiled, not run on hardware
//
#include <hip/hip_runtime.h>

// ---------------------------------------------------------------------------
// GRU scan (M=128 steps) for MI455X / gfx1250, wave32 + WMMA bf16.
// Per step one fused kernel computes both GEMMs (emb@W_ih^T, h@W_hh^T) with
// v_wmma_f32_16x16x32_bf16 (fp32 accumulate), applies gates, masked update.
// State ping-pongs in workspace (fp32 + bf16 mirror). The bf16 state panel is
// staged into LDS with gfx1250 async global->LDS copies (ASYNCcnt).
// ---------------------------------------------------------------------------

#define MLEN 128
#define NB   1024
#define EDIM 512
#define LDA  520            // LDS row stride in shorts (pad breaks bank conflicts)
#define GATE (EDIM * EDIM)  // gate stride in shorts within [3E,E] weight array

typedef __bf16 bf16;
typedef bf16  v16bf __attribute__((ext_vector_type(16)));
typedef bf16  v8bf  __attribute__((ext_vector_type(8)));
typedef float v8f   __attribute__((ext_vector_type(8)));
typedef int   v4i   __attribute__((ext_vector_type(4)));

#if __has_builtin(__builtin_amdgcn_global_load_async_to_lds_b128)
#define HAVE_ASYNC_COPY 1
typedef __attribute__((address_space(1))) v4i glb_v4i;   // global int4
typedef __attribute__((address_space(3))) v4i lds_v4i;   // LDS int4
#else
#define HAVE_ASYNC_COPY 0
#endif

__device__ __forceinline__ unsigned short f2bf(float f) {
    unsigned int u = __float_as_uint(f);               // RNE fp32 -> bf16
    unsigned int r = u + 0x7FFFu + ((u >> 16) & 1u);
    return (unsigned short)(r >> 16);
}

__device__ __forceinline__ float sigmoidf(float x) {
    return 1.0f / (1.0f + __expf(-x));
}

__device__ __forceinline__ v16bf cat8(v8bf lo, v8bf hi) {
    return __builtin_shufflevector(lo, hi, 0,1,2,3,4,5,6,7,8,9,10,11,12,13,14,15);
}

__device__ __forceinline__ v8f wmma_bf16(v16bf a, v16bf b, v8f c) {
    return __builtin_amdgcn_wmma_f32_16x16x32_bf16(
        false, a, false, b, (short)0, c, false, false);
}

// --------------------------- init / convert kernels ------------------------

__global__ void gru_init_kernel(float* __restrict__ st0,
                                unsigned short* __restrict__ hb0,
                                int* __restrict__ alive) {
    int i = blockIdx.x * blockDim.x + threadIdx.x;
    if (i < NB * EDIM) { st0[i] = 0.0f; hb0[i] = 0; }
    if (i < NB) alive[i] = 1;
}

__global__ void gru_cvt_kernel(const float* __restrict__ wih_f,
                               const float* __restrict__ whh_f,
                               unsigned short* __restrict__ wih,
                               unsigned short* __restrict__ whh) {
    int i = blockIdx.x * blockDim.x + threadIdx.x;
    if (i < 3 * EDIM * EDIM) {
        wih[i] = f2bf(wih_f[i]);
        whh[i] = f2bf(whh_f[i]);
    }
}

__global__ void gru_alive_kernel(const int* __restrict__ utt,
                                 int* __restrict__ alive,
                                 const int* __restrict__ term, int t) {
    int n = blockIdx.x * blockDim.x + threadIdx.x;
    if (n < NB && utt[t * NB + n] == *term) alive[n] = 0;
}

// ------------------------------- step kernel -------------------------------
// grid = (NB/32, EDIM/64), block = 256 threads = 8 wave32.
// Tile: 32 rows x 64 cols of E. Each wave owns one 16x16 output subtile with
// 4 fp32 accumulators: r (gi+gh), z (gi+gh), i_n, h_n.

__global__ __launch_bounds__(256)
void gru_step_kernel(const int* __restrict__ utt,
                     const float* __restrict__ emb_table,
                     const unsigned short* __restrict__ wih,   // [3E,E] bf16
                     const unsigned short* __restrict__ whh,   // [3E,E] bf16
                     const float* __restrict__ b_ih,
                     const float* __restrict__ b_hh,
                     const int* __restrict__ alive,
                     const float* __restrict__ h_src,          // fp32 state
                     const unsigned short* __restrict__ hb_src,// bf16 state
                     float* __restrict__ h_dst,
                     unsigned short* __restrict__ hb_dst,
                     int t) {
    __shared__ __align__(16) unsigned short ldsE[32 * LDA];  // emb panel
    __shared__ __align__(16) unsigned short ldsS[32 * LDA];  // state panel

    const int tid = threadIdx.x;
    const int r0  = blockIdx.x * 32;
    const int c0  = blockIdx.y * 64;

    // ---- stage A panels: 8 threads per row, 64 cols (128B) each ----
    {
        const int row  = tid >> 3;          // 0..31
        const int cs   = (tid & 7) * 64;    // shorts
        const int grow = r0 + row;

        // state panel: bf16 mirror -> LDS.  gfx1250 async DMA path when
        // available (global_load_async_to_lds_b128, tracked by ASYNCcnt).
#if HAVE_ASYNC_COPY
        {
            glb_v4i* gsrc = (glb_v4i*)(hb_src + (size_t)grow * EDIM + cs);
            lds_v4i* ldst = (lds_v4i*)&ldsS[row * LDA + cs];
            #pragma unroll
            for (int i = 0; i < 8; ++i)   // 8 x 16B per thread
                __builtin_amdgcn_global_load_async_to_lds_b128(
                    gsrc + i, ldst + i, 0, 0);
        }
#else
        {
            const uint4* gsrc = (const uint4*)(hb_src + (size_t)grow * EDIM + cs);
            uint4* ldst = (uint4*)&ldsS[row * LDA + cs];
            #pragma unroll
            for (int i = 0; i < 8; ++i) ldst[i] = gsrc[i];
        }
#endif

        // emb panel: token gather + fp32 -> bf16 convert
        const int tok = utt[t * NB + grow];
        const float* esrc = emb_table + (size_t)tok * EDIM + cs;
        unsigned short* edst = &ldsE[row * LDA + cs];
        #pragma unroll
        for (int i = 0; i < 64; i += 4) {
            float4 ev = *(const float4*)(esrc + i);
            edst[i + 0] = f2bf(ev.x); edst[i + 1] = f2bf(ev.y);
            edst[i + 2] = f2bf(ev.z); edst[i + 3] = f2bf(ev.w);
        }
    }
#if HAVE_ASYNC_COPY
#if __has_builtin(__builtin_amdgcn_s_wait_asynccnt)
    __builtin_amdgcn_s_wait_asynccnt(0);
#else
    asm volatile("s_wait_asynccnt 0" ::: "memory");
#endif
#endif
    __syncthreads();

    const int lane = tid & 31;
    const int wv   = tid >> 5;            // 0..7
    const int rs   = (wv & 1) * 16;       // row subtile base within tile
    const int csb  = (wv >> 1) * 16;      // col subtile base within tile
    const int lj   = lane & 15;
    const int lhi  = lane >> 4;           // lane half

    v8f acc_r = {}, acc_z = {}, acc_in = {}, acc_hn = {};

    // A fragments (contiguous K order, matching B):
    //   half 0: K = kc + {0..15}; half 1: K = kc + {16..31}
    const int arow = rs + lj;
    const unsigned short* aE = &ldsE[arow * LDA + lhi * 16];
    const unsigned short* aS = &ldsS[arow * LDA + lhi * 16];

    // B fragments: 2 per-lane bases; gate stride + kc folded into the 24-bit
    // immediate offset of global_load_b128 by the full unroll below.
    const int wcol = c0 + csb + lj;
    const unsigned short* bih0 = wih + (size_t)wcol * EDIM + lhi * 16;
    const unsigned short* bhh0 = whh + (size_t)wcol * EDIM + lhi * 16;

    #pragma unroll
    for (int g = 0; g < 3; ++g) {          // gfx1250 global_prefetch_b8
        __builtin_prefetch(bih0 + g * GATE, 0, 3);
        __builtin_prefetch(bhh0 + g * GATE, 0, 3);
    }

    #pragma unroll
    for (int kc = 0; kc < EDIM; kc += 32) {
        v16bf Ae = cat8(*(const v8bf*)(aE + kc), *(const v8bf*)(aE + kc + 8));
        v16bf As = cat8(*(const v8bf*)(aS + kc), *(const v8bf*)(aS + kc + 8));

        v16bf Bihr = *(const v16bf*)(bih0 + 0 * GATE + kc);
        v16bf Bihz = *(const v16bf*)(bih0 + 1 * GATE + kc);
        v16bf Bihn = *(const v16bf*)(bih0 + 2 * GATE + kc);
        v16bf Bhhr = *(const v16bf*)(bhh0 + 0 * GATE + kc);
        v16bf Bhhz = *(const v16bf*)(bhh0 + 1 * GATE + kc);
        v16bf Bhhn = *(const v16bf*)(bhh0 + 2 * GATE + kc);

        acc_r  = wmma_bf16(Ae, Bihr, acc_r);   // r: gi + gh summed in acc
        acc_r  = wmma_bf16(As, Bhhr, acc_r);
        acc_z  = wmma_bf16(Ae, Bihz, acc_z);   // z: gi + gh summed in acc
        acc_z  = wmma_bf16(As, Bhhz, acc_z);
        acc_in = wmma_bf16(Ae, Bihn, acc_in);  // n: i_n, h_n kept separate
        acc_hn = wmma_bf16(As, Bhhn, acc_hn);
    }

    // ---- epilogue: biases, gates, alive-masked state update ----
    const int cg = c0 + csb + lj;
    const float br  = b_ih[cg] + b_hh[cg];
    const float bz  = b_ih[EDIM + cg] + b_hh[EDIM + cg];
    const float bin = b_ih[2 * EDIM + cg];
    const float bhn = b_hh[2 * EDIM + cg];

    const int rowbase = r0 + rs + lhi * 8;   // C/D layout: M = v + 8*lane_half
    #pragma unroll
    for (int v = 0; v < 8; ++v) {
        const int row = rowbase + v;
        const size_t idx = (size_t)row * EDIM + cg;
        const float hold = h_src[idx];
        const float r = sigmoidf(acc_r[v] + br);
        const float z = sigmoidf(acc_z[v] + bz);
        const float n = tanhf(acc_in[v] + bin + r * (acc_hn[v] + bhn));
        const float hnew = (1.0f - z) * n + z * hold;
        const float out  = alive[row] ? hnew : hold;
        h_dst[idx]  = out;
        hb_dst[idx] = f2bf(out);             // bf16 mirror for next step's A
    }
}

// ------------------------------- launcher ----------------------------------

extern "C" void kernel_launch(void* const* d_in, const int* in_sizes, int n_in,
                              void* d_out, int out_size, void* d_ws, size_t ws_size,
                              hipStream_t stream) {
    const int*   utt  = (const int*)d_in[0];     // [M,N] int32
    const float* emb  = (const float*)d_in[1];   // [V,E] f32
    const float* Wih  = (const float*)d_in[2];   // [3E,E] f32
    const float* Whh  = (const float*)d_in[3];   // [3E,E] f32
    const float* bih  = (const float*)d_in[4];   // [3E] f32
    const float* bhh  = (const float*)d_in[5];   // [3E] f32
    const int*   term = (const int*)d_in[6];     // scalar

    // ws layout: fp32 state ping-pong, alive, bf16 weights, bf16 state mirror
    float* st0 = (float*)d_ws;
    float* st1 = st0 + (size_t)NB * EDIM;
    int*   alive = (int*)(st1 + (size_t)NB * EDIM);
    unsigned short* wih16 = (unsigned short*)(alive + NB);
    unsigned short* whh16 = wih16 + (size_t)3 * EDIM * EDIM;
    unsigned short* hb0   = whh16 + (size_t)3 * EDIM * EDIM;
    unsigned short* hb1   = hb0 + (size_t)NB * EDIM;

    gru_init_kernel<<<(NB * EDIM + 255) / 256, 256, 0, stream>>>(st0, hb0, alive);
    gru_cvt_kernel<<<(3 * EDIM * EDIM + 255) / 256, 256, 0, stream>>>(Wih, Whh, wih16, whh16);

    float* src = st0;           float* dst = st1;
    unsigned short* bsrc = hb0; unsigned short* bdst = hb1;
    for (int t = 0; t < MLEN; ++t) {
        dim3 grid(NB / 32, EDIM / 64);
        gru_step_kernel<<<grid, 256, 0, stream>>>(utt, emb, wih16, whh16,
                                                  bih, bhh, alive,
                                                  src, bsrc, dst, bdst, t);
        gru_alive_kernel<<<(NB + 255) / 256, 256, 0, stream>>>(utt, alive, term, t);
        float* tf = src; src = dst; dst = tf;
        unsigned short* tb = bsrc; bsrc = bdst; bdst = tb;
    }

    (void)hipMemcpyAsync(d_out, src, (size_t)NB * EDIM * sizeof(float),
                         hipMemcpyDeviceToDevice, stream);
}
